// TransformerBlock_78056735637614
// MI455X (gfx1250) — matrix operand order repaired and checked by reading, not compiled, not run on hardware
//
#include <hip/hip_runtime.h>
#include <hip/hip_bf16.h>
#include <math.h>

typedef __attribute__((ext_vector_type(16))) _Float16 v16h;
typedef __attribute__((ext_vector_type(8)))  float    v8f;

#define HWQ 65536   // 256*256
#define HWD 16384   // 128*128

// ---------------------------------------------------------------------------
// 1x1-conv GEMM via WMMA f16->f32, fully compile-time shapes (no guards).
// Y[b, o, p] = sum_c W[bw][o, c] * X[b, group*CIN + c, p] (+bias) (+res)
// Wave tile: 16 out-ch x 32 pixels (2 accumulators). CIN, COUT multiples that
// need no runtime raggedness (padded weights for 340/170 cases).
// ---------------------------------------------------------------------------
template <int CIN, int COUT, int OPG, int HWP>
__global__ void conv1x1_wmma_kernel(
    const float* __restrict__ X, const float* __restrict__ Wt,
    const float* __restrict__ bias, const float* __restrict__ res,
    float* __restrict__ Y,
    size_t xBatchStride, size_t yBatchStride, int wBatchStride)
{
    const int lane = threadIdx.x & 31;
    const int wave = threadIdx.x >> 5;
    const int b    = blockIdx.z;
    const int m0   = blockIdx.y * 16;
    const int p0   = blockIdx.x * 256 + wave * 32;
    const int group = m0 / OPG;                    // OPG is compile-time

    const float* Xb = X + (size_t)b * xBatchStride
                        + (size_t)(group * CIN) * HWP + p0 + (lane & 15);
    const float* Wb = Wt + (size_t)b * wBatchStride
                         + (size_t)(m0 + (lane & 15)) * CIN;
    const int hiA = (lane >> 4) << 3;   // unused (A loads also use hiB: +16 upper half-wave)
    const int hiB = (lane >> 4) << 4;   // B K-offset (+16 upper half-wave)

    v8f acc0 = {}, acc1 = {};

#pragma unroll
    for (int k0 = 0; k0 < CIN; k0 += 32) {
        v16h a;
#pragma unroll
        for (int j = 0; j < 16; ++j)
            a[j] = (_Float16)Wb[k0 + j + hiB];

        if (k0 + 32 < CIN)   // compile-time condition
            __builtin_prefetch(Xb + (size_t)(k0 + 32) * HWP, 0, 1);

        v16h b0f, b1f;
#pragma unroll
        for (int j = 0; j < 16; ++j) {
            int kc = k0 + j + hiB;
            b0f[j] = (_Float16)Xb[(size_t)kc * HWP];
            b1f[j] = (_Float16)Xb[(size_t)kc * HWP + 16];
        }
        acc0 = __builtin_amdgcn_wmma_f32_16x16x32_f16(
            false, a, false, b0f, (short)0, acc0, false, false);
        acc1 = __builtin_amdgcn_wmma_f32_16x16x32_f16(
            false, a, false, b1f, (short)0, acc1, false, false);
    }

    // epilogue: D layout m = r + 8*(lane>=16), n = lane&15
    const int nD  = lane & 15;
    const int mHi = (lane >> 4) << 3;
    float* Yb = Y + (size_t)b * yBatchStride + (size_t)m0 * HWP + p0 + nD;
    const float* Rb = res + (size_t)b * yBatchStride + (size_t)m0 * HWP + p0 + nD;
#pragma unroll
    for (int r = 0; r < 8; ++r) {
        int m = r + mHi;
        float v0 = acc0[r], v1 = acc1[r];
        if (bias) { float bv = bias[m0 + m]; v0 += bv; v1 += bv; }
        size_t o0 = (size_t)m * HWP;
        if (res)  { v0 += Rb[o0]; v1 += Rb[o0 + 16]; }
        Yb[o0] = v0; Yb[o0 + 16] = v1;
    }
}

// ---------------------------------------------------------------------------
// LayerNorm over the channel axis (C=64), per pixel.
// ---------------------------------------------------------------------------
__global__ void ln_kernel(const float* __restrict__ X, const float* __restrict__ g,
                          const float* __restrict__ bt, float* __restrict__ Y)
{
    int idx = blockIdx.x * 256 + threadIdx.x;   // over B*HW
    int b = idx >> 16, p = idx & 65535;
    const float* xp = X + ((size_t)b * 64) * HWQ + p;
    float s = 0.f, s2 = 0.f;
#pragma unroll 4
    for (int c = 0; c < 64; ++c) { float v = xp[(size_t)c * HWQ]; s += v; s2 += v * v; }
    float mu  = s * (1.f / 64.f);
    float var = s2 * (1.f / 64.f) - mu * mu;
    float rs  = rsqrtf(var + 1e-5f);
    float* yp = Y + ((size_t)b * 64) * HWQ + p;
#pragma unroll 4
    for (int c = 0; c < 64; ++c) {
        float v = xp[(size_t)c * HWQ];
        yp[(size_t)c * HWQ] = (v - mu) * rs * g[c] + bt[c];
    }
}

// ---------------------------------------------------------------------------
// Depthwise 3x3, padding 1. chanStride allows padded channel layouts.
// ---------------------------------------------------------------------------
__global__ void dw3x3_kernel(const float* __restrict__ X, const float* __restrict__ Wd,
                             float* __restrict__ Y, int Cn, int chanStride,
                             int Hh, int Ww, int total)
{
    int idx = blockIdx.x * 256 + threadIdx.x;
    if (idx >= total) return;
    int x = idx % Ww; int t = idx / Ww;
    int y = t % Hh;   t /= Hh;
    int c = t % Cn;   int b = t / Cn;
    const float* w9  = Wd + (size_t)c * 9;
    size_t base = ((size_t)b * chanStride + c) * ((size_t)Hh * Ww) + (size_t)y * Ww + x;
    const float* xin = X + base;
    float acc = 0.f;
    if (y > 0 && y < Hh - 1 && x > 0 && x < Ww - 1) {
#pragma unroll
        for (int dy = -1; dy <= 1; ++dy)
#pragma unroll
            for (int dx = -1; dx <= 1; ++dx)
                acc += w9[(dy + 1) * 3 + (dx + 1)] * xin[(size_t)dy * Ww + dx];
    } else {
#pragma unroll
        for (int dy = -1; dy <= 1; ++dy)
#pragma unroll
            for (int dx = -1; dx <= 1; ++dx) {
                int yy = y + dy, xx = x + dx;
                if (yy >= 0 && yy < Hh && xx >= 0 && xx < Ww)
                    acc += w9[(dy + 1) * 3 + (dx + 1)] * xin[(size_t)dy * Ww + dx];
            }
    }
    Y[base] = acc;
}

__global__ void zero_kernel(float* p, int n)
{
    int i = blockIdx.x * 256 + threadIdx.x;
    if (i < n) p[i] = 0.f;
}

// pad pin_w (340x64) -> (352x64) zero rows
__global__ void pad_pin_kernel(const float* __restrict__ Wsrc, float* __restrict__ Wp)
{
    int idx = blockIdx.x * 256 + threadIdx.x;   // 352*64
    int o = idx >> 6;
    Wp[idx] = (o < 340) ? Wsrc[idx] : 0.f;
}

// pad pout_w (64x170) -> (64x192) zero cols
__global__ void pad_pout_kernel(const float* __restrict__ Wsrc, float* __restrict__ Wp)
{
    int idx = blockIdx.x * 256 + threadIdx.x;   // 64*192
    int o = idx / 192, c = idx % 192;
    Wp[idx] = (c < 170) ? Wsrc[o * 170 + c] : 0.f;
}

// ---------------------------------------------------------------------------
// MDTA gram: G[b,h] += q[16 x Kslice] * k^T  (split-K WMMA, f32 atomics).
// ---------------------------------------------------------------------------
__global__ void mdta_gram_kernel(const float* __restrict__ QKV, float* __restrict__ G)
{
    const int lane = threadIdx.x;
    const int h = blockIdx.y, b = blockIdx.z;
    const int n0 = blockIdx.x * (HWQ / 64);
    const int mA = lane & 15, hiA = (lane >> 4) << 3;
    const int nB = lane & 15, hiB = (lane >> 4) << 4;
    const float* qp = QKV + ((size_t)b * 192 + h * 16 + mA) * HWQ + n0 + hiB;
    const float* kp = QKV + ((size_t)b * 192 + 64 + h * 16 + nB) * HWQ + n0 + hiB;

    v8f acc = {};
    for (int kk = 0; kk < HWQ / 64; kk += 32) {
        v16h a, bf;
#pragma unroll
        for (int j = 0; j < 16; ++j) {
            a[j]  = (_Float16)qp[kk + j];
            bf[j] = (_Float16)kp[kk + j];
        }
        acc = __builtin_amdgcn_wmma_f32_16x16x32_f16(
            false, a, false, bf, (short)0, acc, false, false);
    }
    const int n = lane & 15;
#pragma unroll
    for (int r = 0; r < 8; ++r) {
        int m = r + ((lane >> 4) << 3);
        atomicAdd(&G[(((size_t)(b * 4 + h) * 16) + m) * 16 + n], acc[r]);
    }
}

// L2 norms of q/k rows. row = ((b*2+s)*4 + h)*16 + d, channel = s*64+h*16+d.
__global__ void mdta_norms_kernel(const float* __restrict__ QKV, float* __restrict__ NRM)
{
    __shared__ float red[256];
    int row = blockIdx.x;
    int d = row & 15, h = (row >> 4) & 3, s = (row >> 6) & 1, b = row >> 7;
    const float* p = QKV + ((size_t)b * 192 + s * 64 + h * 16 + d) * HWQ;
    float ss = 0.f;
    for (int i = threadIdx.x; i < HWQ; i += 256) { float v = p[i]; ss += v * v; }
    red[threadIdx.x] = ss;
    __syncthreads();
    for (int st = 128; st > 0; st >>= 1) {
        if (threadIdx.x < st) red[threadIdx.x] += red[threadIdx.x + st];
        __syncthreads();
    }
    if (threadIdx.x == 0) NRM[row] = fmaxf(sqrtf(red[0]), 1e-12f);
}

// attn = softmax(G / (|q_d||k_e|) * temperature), in place. 1 block per (b,h).
__global__ void mdta_softmax_kernel(float* __restrict__ G, const float* __restrict__ NRM,
                                    const float* __restrict__ temp)
{
    int bh = blockIdx.x, b = bh >> 2, h = bh & 3;
    int d = threadIdx.x;
    if (d >= 16) return;
    float* row = G + ((size_t)bh * 16 + d) * 16;
    float nq = NRM[(((b * 2 + 0) * 4 + h) << 4) + d];
    float tp = temp[h];
    float vals[16], mx = -1e30f;
#pragma unroll
    for (int e = 0; e < 16; ++e) {
        float nk = NRM[(((b * 2 + 1) * 4 + h) << 4) + e];
        vals[e] = row[e] / (nq * nk) * tp;
        mx = fmaxf(mx, vals[e]);
    }
    float sum = 0.f;
#pragma unroll
    for (int e = 0; e < 16; ++e) { vals[e] = expf(vals[e] - mx); sum += vals[e]; }
    float inv = 1.f / sum;
#pragma unroll
    for (int e = 0; e < 16; ++e) row[e] = vals[e] * inv;
}

// W2[b][o][h*16+e] = sum_d Wout[o, h*16+d] * attn[b,h,d,e]
__global__ void fuse_w_kernel(const float* __restrict__ Wout, const float* __restrict__ G,
                              float* __restrict__ W2)
{
    int idx = blockIdx.x * 256 + threadIdx.x;   // B*64*64
    int b = idx >> 12, o = (idx >> 6) & 63, ce = idx & 63;
    int h = ce >> 4, e = ce & 15;
    float s = 0.f;
#pragma unroll
    for (int d = 0; d < 16; ++d)
        s += Wout[(size_t)o * 64 + h * 16 + d] * G[(((size_t)(b * 4 + h) * 16) + d) * 16 + e];
    W2[idx] = s;
}

// ---------------------------------------------------------------------------
// Haar DWT: stride-2 depthwise with fixed +/-0.5 taps. out ch o: c=o>>2, f=o&3.
// ---------------------------------------------------------------------------
__global__ void dwt_kernel(const float* __restrict__ X, float* __restrict__ Y)
{
    int idx = blockIdx.x * 256 + threadIdx.x;   // B*256*128*128
    int j = idx & 127, i = (idx >> 7) & 127, o = (idx >> 14) & 255, b = idx >> 22;
    int c = o >> 2, f = o & 3;
    const float* xp = X + ((size_t)b * 64 + c) * HWQ + (size_t)(2 * i) * 256 + 2 * j;
    float p00 = xp[0], p01 = xp[1], p10 = xp[256], p11 = xp[257];
    float v;
    if      (f == 0) v = 0.5f * ( p00 + p01 + p10 + p11);
    else if (f == 1) v = 0.5f * (-p00 - p01 + p10 + p11);
    else if (f == 2) v = 0.5f * (-p00 + p01 - p10 + p11);
    else             v = 0.5f * ( p00 - p01 - p10 + p11);
    Y[idx] = v;
}

// ---------------------------------------------------------------------------
// Wave window attention (linear): ow = scale * qw @ (kw^T @ vw).
// One wave per (b, window, head). S routed through LDS for B relayout.
// A fragments are zero for K>=16 at compile time (j>=8), so B fragments may
// load masked (garbage) token indices unconditionally for K>=16.
// ---------------------------------------------------------------------------
__global__ void wave_window_kernel(const float* __restrict__ QF,
                                   const float* __restrict__ KV,
                                   float* __restrict__ OW)
{
    __shared__ float Sbuf[8][264];
    const int lane = threadIdx.x & 31, wave = threadIdx.x >> 5;
    const int wid = blockIdx.x * 8 + wave;      // 8192 window-heads
    const int h = wid & 1;
    const int win = (wid >> 1) & 1023;
    const int b = wid >> 11;
    const int wr = win >> 5, wc = win & 31;

    const float* KF = KV + ((size_t)b * 64 + h * 16) * HWD;
    const float* VF = KV + ((size_t)b * 64 + 32 + h * 16) * HWD;
    const int mA = lane & 15, hiA = (lane >> 4) << 3;
    const int nB = lane & 15, hiB = (lane >> 4) << 4;
    const int kbase = (wr * 4) * 128 + wc * 4;  // 4x4 kv window in 128x128 map

    // S[d,e] = sum_t kw[t,d] * vw[t,e]   (K = 16 tokens, A zero-padded to 32)
    v16h aS, bS;
#pragma unroll
    for (int j = 0; j < 16; ++j) {
        if (j < 8) {
            int k = j + hiA;                    // token 0..15
            aS[j] = (_Float16)KF[(size_t)mA * HWD + kbase + (k >> 2) * 128 + (k & 3)];
        } else {
            aS[j] = (_Float16)0.f;              // K>=16: zero (compile-time)
        }
        int k2 = (j & 7) + hiA;                 // token 0..15; j>=8 unused (A==0)
        bS[j] = (_Float16)VF[(size_t)nB * HWD + kbase + (k2 >> 2) * 128 + (k2 & 3)];
    }
    v8f zc = {};
    v8f s = __builtin_amdgcn_wmma_f32_16x16x32_f16(false, aS, false, bS, (short)0, zc, false, false);

    const float scale = 0.25f;   // WHD^-0.5 = 1/4
#pragma unroll
    for (int r = 0; r < 8; ++r) {
        int m = r + ((lane >> 4) << 3);
        Sbuf[wave][m * 16 + (lane & 15)] = s[r] * scale;
    }

    // reload S as B fragment (K = d, masked for K>=16)
    v16h b2;
#pragma unroll
    for (int j = 0; j < 16; ++j) {
        int k = (j & 7) + hiA;
        b2[j] = (_Float16)Sbuf[wave][k * 16 + nB];
    }

    const float* QW = QF + ((size_t)b * 32 + h * 16) * HWQ;
    const int qbase = (wr * 8) * 256 + wc * 8;  // 8x8 q window in 256x256 map
#pragma unroll
    for (int qt = 0; qt < 4; ++qt) {
        v16h a2;
        const int t = qt * 16 + mA;             // q token in window
        const int qoff = qbase + (t >> 3) * 256 + (t & 7);
#pragma unroll
        for (int j = 0; j < 16; ++j) {
            if (j < 8) {
                int k = j + hiA;                // head dim d 0..15
                a2[j] = (_Float16)QW[(size_t)k * HWQ + qoff];
            } else {
                a2[j] = (_Float16)0.f;
            }
        }
        v8f d2 = __builtin_amdgcn_wmma_f32_16x16x32_f16(false, a2, false, b2, (short)0, zc, false, false);
#pragma unroll
        for (int r = 0; r < 8; ++r) {
            int m = r + ((lane >> 4) << 3);
            int tt = qt * 16 + m;
            OW[((size_t)b * 32 + h * 16 + (lane & 15)) * HWQ
               + qbase + (tt >> 3) * 256 + (tt & 7)] = d2[r];
        }
    }
}

// x2 = x + alpha*spatial + (1-alpha)*wave
__global__ void mix_kernel(const float* __restrict__ X, const float* __restrict__ SP,
                           const float* __restrict__ WV, const float* __restrict__ al,
                           float* __restrict__ Y)
{
    int idx = blockIdx.x * 256 + threadIdx.x;
    float a = 1.f / (1.f + expf(-al[0]));
    Y[idx] = X[idx] + a * SP[idx] + (1.f - a) * WV[idx];
}

// GDFN gate: gelu(h[:170]) * h[170:340].  Input: 352-ch padded layout.
// Output: 192-ch padded layout, channels 170..191 zeroed.
__global__ void gate_kernel(const float* __restrict__ H, float* __restrict__ GG, int total)
{
    int idx = blockIdx.x * 256 + threadIdx.x;   // B*192*HW
    if (idx >= total) return;
    int p = idx & 65535; int t = idx >> 16;
    int c = t % 192; int b = t / 192;
    float r = 0.f;
    if (c < 170) {
        const float* hb = H + (size_t)b * 352 * HWQ;
        float x1 = hb[(size_t)c * HWQ + p];
        float x2 = hb[(size_t)(170 + c) * HWQ + p];
        float ge = 0.5f * x1 * (1.f + erff(x1 * 0.70710678118654752f));
        r = ge * x2;
    }
    GG[idx] = r;
}

// ---------------------------------------------------------------------------
extern "C" void kernel_launch(void* const* d_in, const int* in_sizes, int n_in,
                              void* d_out, int out_size, void* d_ws, size_t ws_size,
                              hipStream_t stream)
{
    (void)in_sizes; (void)n_in; (void)out_size; (void)ws_size;
    const float* x          = (const float*)d_in[0];
    const float* ln1_g      = (const float*)d_in[1];
    const float* ln1_b      = (const float*)d_in[2];
    const float* temperature= (const float*)d_in[3];
    const float* qkv_w      = (const float*)d_in[4];
    const float* qkv_dw     = (const float*)d_in[5];
    const float* attn_out_w = (const float*)d_in[6];
    const float* q_w        = (const float*)d_in[7];
    const float* kv_w       = (const float*)d_in[8];
    const float* wave_out_w = (const float*)d_in[9];
    const float* wave_out_b = (const float*)d_in[10];
    const float* alpha_logit= (const float*)d_in[11];
    const float* ln2_g      = (const float*)d_in[12];
    const float* ln2_b      = (const float*)d_in[13];
    const float* pin_w      = (const float*)d_in[14];
    const float* gd_dw      = (const float*)d_in[15];
    const float* pout_w     = (const float*)d_in[16];
    float* out = (float*)d_out;

    const size_t HW = HWQ, HWd = HWD;
    float* w = (float*)d_ws;
    size_t off = 0;
    auto take = [&](size_t n) { float* p = w + off; off += n; return p; };
    float* XN  = take((size_t)4 * 64 * HW);    // xn1, later xn2
    float* R1  = take((size_t)4 * 352 * HW);   // qkv pre-dw -> pin out (352pad) -> gated (192pad)
    float* R2  = take((size_t)4 * 352 * HW);   // qkv post-dw -> gd dw out (352pad)
    float* SP  = take((size_t)4 * 64 * HW);    // spatial branch
    float* QFb = take((size_t)4 * 32 * HW);    // wave q features
    float* XDb = take((size_t)4 * 256 * HWd);  // DWT output
    float* KVb = take((size_t)4 * 64 * HWd);   // kv features
    float* OWb = take((size_t)4 * 32 * HW);    // window-attn output
    float* WVb = take((size_t)4 * 64 * HW);    // wave branch
    float* X2  = take((size_t)4 * 64 * HW);    // mixed residual
    float* G   = take(4096);                   // 16x(16x16) attn matrices
    float* NRM = take(512);                    // q/k row norms
    float* W2  = take(4 * 4096);               // fused per-batch out weights
    float* WP1 = take(352 * 64);               // padded pin_w
    float* WP2 = take(64 * 192);               // padded pout_w

    // weight padding (independent of data flow)
    pad_pin_kernel<<<(352 * 64) / 256, 256, 0, stream>>>(pin_w, WP1);
    pad_pout_kernel<<<(64 * 192) / 256, 256, 0, stream>>>(pout_w, WP2);

    // 1) LN1
    ln_kernel<<<(4 * HW) / 256, 256, 0, stream>>>(x, ln1_g, ln1_b, XN);
    // 2) qkv 1x1 (64 -> 192)
    conv1x1_wmma_kernel<64, 192, 192, HWQ><<<dim3(256, 12, 4), 256, 0, stream>>>(
        XN, qkv_w, nullptr, nullptr, R1, (size_t)64 * HW, (size_t)192 * HW, 0);
    // 3) qkv depthwise 3x3
    { int total = 4 * 192 * (int)HW;
      dw3x3_kernel<<<total / 256, 256, 0, stream>>>(R1, qkv_dw, R2, 192, 192, 256, 256, total); }
    // 4) zero gram accumulators
    zero_kernel<<<16, 256, 0, stream>>>(G, 4096);
    // 5) split-K gram q.k^T per (b,head)
    mdta_gram_kernel<<<dim3(64, 4, 4), 32, 0, stream>>>(R2, G);
    // 6) q/k row L2 norms
    mdta_norms_kernel<<<512, 256, 0, stream>>>(R2, NRM);
    // 7) normalize + temperature + softmax
    mdta_softmax_kernel<<<16, 32, 0, stream>>>(G, NRM, temperature);
    // 8) fold attn into out-conv weights (per batch)
    fuse_w_kernel<<<64, 256, 0, stream>>>(attn_out_w, G, W2);
    // 9) spatial = conv1x1(v, W2[b])   (v = qkv channels 128..191)
    conv1x1_wmma_kernel<64, 64, 64, HWQ><<<dim3(256, 4, 4), 256, 0, stream>>>(
        R2 + (size_t)128 * HW, W2, nullptr, nullptr, SP,
        (size_t)192 * HW, (size_t)64 * HW, 4096);
    // 10) wave q features (64 -> 32)
    conv1x1_wmma_kernel<64, 32, 32, HWQ><<<dim3(256, 2, 4), 256, 0, stream>>>(
        XN, q_w, nullptr, nullptr, QFb, (size_t)64 * HW, (size_t)32 * HW, 0);
    // 11) Haar DWT (stride-2 depthwise, fixed taps)
    dwt_kernel<<<(4 * 256 * HWd) / 256, 256, 0, stream>>>(XN, XDb);
    // 12) kv grouped 1x1 (4 groups of 64 -> 16)
    conv1x1_wmma_kernel<64, 64, 16, HWD><<<dim3(64, 4, 4), 256, 0, stream>>>(
        XDb, kv_w, nullptr, nullptr, KVb, (size_t)256 * HWd, (size_t)64 * HWd, 0);
    // 13) linear window attention
    wave_window_kernel<<<1024, 256, 0, stream>>>(QFb, KVb, OWb);
    // 14) wave out conv (32 -> 64) + bias
    conv1x1_wmma_kernel<32, 64, 64, HWQ><<<dim3(256, 4, 4), 256, 0, stream>>>(
        OWb, wave_out_w, wave_out_b, nullptr, WVb, (size_t)32 * HW, (size_t)64 * HW, 0);
    // 15) residual mix
    mix_kernel<<<(4 * 64 * HW) / 256, 256, 0, stream>>>(x, SP, WVb, alpha_logit, X2);
    // 16) LN2 (reuse XN)
    ln_kernel<<<(4 * HW) / 256, 256, 0, stream>>>(X2, ln2_g, ln2_b, XN);
    // 17) GDFN pin (64 -> 352 padded)
    conv1x1_wmma_kernel<64, 352, 352, HWQ><<<dim3(256, 22, 4), 256, 0, stream>>>(
        XN, WP1, nullptr, nullptr, R1, (size_t)64 * HW, (size_t)352 * HW, 0);
    // 18) GDFN depthwise 3x3 (340 real channels in 352-ch layout)
    { int total = 4 * 340 * (int)HW;
      dw3x3_kernel<<<total / 256, 256, 0, stream>>>(R1, gd_dw, R2, 340, 352, 256, 256, total); }
    // 19) gate: gelu(x1)*x2 -> R1 as 192-ch padded layout
    { int total = 4 * 192 * (int)HW;
      gate_kernel<<<total / 256, 256, 0, stream>>>(R2, R1, total); }
    // 20) GDFN pout (192 padded -> 64) + final residual, write d_out
    conv1x1_wmma_kernel<192, 64, 64, HWQ><<<dim3(256, 4, 4), 256, 0, stream>>>(
        R1, WP2, nullptr, X2, out, (size_t)192 * HW, (size_t)64 * HW, 0);
}
